// SP_Decoder_57982058496563
// MI455X (gfx1250) — matrix operand order repaired and checked by reading, not compiled, not run on hardware
//
#include <hip/hip_runtime.h>
#include <hip/hip_bf16.h>
#include <math.h>

// ---------------------------------------------------------------------------
// CDNA5 (gfx1250) implementation: every GEMM-shaped op goes through
// v_wmma_f32_16x16x32_bf16 (wave32, 16x16x32 tiles, f32 accumulate).
// LDS layouts are arranged so A/B fragments are contiguous 16B chunks
// (ds_load_b128), per ISA 7.12.2 wave32 fragment layouts.
// ---------------------------------------------------------------------------

typedef __attribute__((ext_vector_type(16))) __bf16 v16bf;
typedef __attribute__((ext_vector_type(8)))  __bf16 v8bf;
typedef __attribute__((ext_vector_type(8)))  float  v8f;

#define TBM 128
#define TBN 128
#define TBK 32
#define KPAD 8   // row stride 40 halves = 80B: 16B-aligned, conflict-free

#define ACT_NONE  0
#define ACT_LRELU 1
#define ACT_TANH  2
#define ACT_DIST  3

#define KNN 10

// 1/sqrt(1 + 1e-5) for bn_eval
#define BN_RSQ 0.99999500003749968f

// ---------------------------------------------------------------------------
// Generic bf16 WMMA GEMM: Y[M,Nd] = act( (A[M,K] * W[Nd,K]^T + bias) * bn )
//   A row-major stride lda, W row-major stride ldw (so B[k][n] = W[n*ldw+k]).
//   ACT_DIST: Y = rowadd[m] + bias[n] - 2*acc   (pairwise sq distances)
// 256 threads = 8 waves; wave grid 4(M) x 2(N); each wave: 2x4 16x16 frags.
// ---------------------------------------------------------------------------
__global__ __launch_bounds__(256) void gemm_bf16_wmma(
    const float* __restrict__ A, int lda,
    const float* __restrict__ W, int ldw,
    float* __restrict__ Y, int ldy,
    int M, int Kd, int Nd,
    const float* __restrict__ bias,
    const float* __restrict__ bng,
    const float* __restrict__ bnb,
    const float* __restrict__ rowadd,
    int act, float alpha)
{
  __shared__ __bf16 As[TBM][TBK + KPAD];   // [m][k]
  __shared__ __bf16 Bs[TBN][TBK + KPAD];   // [n][k]  (transposed tile)

  const int tid  = threadIdx.x;
  const int lane = tid & 31;
  const int wave = tid >> 5;     // 0..7
  const int wm   = wave >> 1;    // 0..3 : 32-row sub-tile
  const int wn   = wave & 1;     // 0..1 : 64-col sub-tile
  const int hf   = lane >> 4;    // wave half (ISA 7.12.2 layouts)
  const int lr   = lane & 15;

  const int m0 = blockIdx.y * TBM;
  const int n0 = blockIdx.x * TBN;

  const bool fastA = (m0 + TBM <= M)  && ((lda & 3) == 0);
  const bool fastB = (n0 + TBN <= Nd) && ((ldw & 3) == 0);

  v8f acc[2][4] = {{{}, {}, {}, {}}, {{}, {}, {}, {}}};

  for (int k0 = 0; k0 < Kd; k0 += TBK) {
    const bool fullK = (k0 + TBK <= Kd);

    // ---- stage A tile (f32 -> bf16) ----
    if (fullK && fastA) {
      // interior: 4x float4 per thread, unguarded -> batched global_load_b128
#pragma unroll
      for (int i = 0; i < 4; ++i) {
        int idx = tid + i * 256;          // 0..1023 over TBM x (TBK/4)
        int r  = idx >> 3;
        int kq = (idx & 7) << 2;
        const float4 v = *(const float4*)(A + (size_t)(m0 + r) * lda + k0 + kq);
        union { __bf16 h[4]; unsigned long long u; } p;
        p.h[0] = (__bf16)v.x; p.h[1] = (__bf16)v.y;
        p.h[2] = (__bf16)v.z; p.h[3] = (__bf16)v.w;
        *(unsigned long long*)&As[r][kq] = p.u;
      }
    } else {
      // edge: branch-free guarded loads (cndmask address, zero result)
      for (int i = tid; i < TBM * (TBK / 2); i += 256) {
        int r  = i >> 4;
        int kp = (i & 15) << 1;
        int gm = m0 + r, gk = k0 + kp;
        bool ok0 = (gm < M) && (gk < Kd);
        bool ok1 = (gm < M) && (gk + 1 < Kd);
        float v0 = A[ok0 ? ((size_t)gm * lda + gk) : 0];
        float v1 = A[ok1 ? ((size_t)gm * lda + gk + 1) : 0];
        if (!ok0) v0 = 0.0f;
        if (!ok1) v1 = 0.0f;
        union { __bf16 h[2]; unsigned u; } p;
        p.h[0] = (__bf16)v0; p.h[1] = (__bf16)v1;
        *(unsigned*)&As[r][kp] = p.u;
      }
    }
    // ---- stage B tile: Bs[n][k] = W[n][k] ----
    if (fullK && fastB) {
#pragma unroll
      for (int i = 0; i < 4; ++i) {
        int idx = tid + i * 256;
        int r  = idx >> 3;
        int kq = (idx & 7) << 2;
        const float4 v = *(const float4*)(W + (size_t)(n0 + r) * ldw + k0 + kq);
        union { __bf16 h[4]; unsigned long long u; } p;
        p.h[0] = (__bf16)v.x; p.h[1] = (__bf16)v.y;
        p.h[2] = (__bf16)v.z; p.h[3] = (__bf16)v.w;
        *(unsigned long long*)&Bs[r][kq] = p.u;
      }
    } else {
      for (int i = tid; i < TBN * (TBK / 2); i += 256) {
        int r  = i >> 4;
        int kp = (i & 15) << 1;
        int gn = n0 + r, gk = k0 + kp;
        bool ok0 = (gn < Nd) && (gk < Kd);
        bool ok1 = (gn < Nd) && (gk + 1 < Kd);
        float v0 = W[ok0 ? ((size_t)gn * ldw + gk) : 0];
        float v1 = W[ok1 ? ((size_t)gn * ldw + gk + 1) : 0];
        if (!ok0) v0 = 0.0f;
        if (!ok1) v1 = 0.0f;
        union { __bf16 h[2]; unsigned u; } p;
        p.h[0] = (__bf16)v0; p.h[1] = (__bf16)v1;
        *(unsigned*)&Bs[r][kp] = p.u;
      }
    }
    // prefetch next K-tile into L2 while we compute (global_prefetch_b8)
    if (k0 + TBK < Kd) {
      int pm = m0 + (tid & (TBM - 1));
      int pn = n0 + (tid & (TBN - 1));
      __builtin_prefetch(&A[(size_t)pm * lda + k0 + TBK], 0, 1);
      __builtin_prefetch(&W[(size_t)pn * ldw + k0 + TBK], 0, 1);
    }
    __syncthreads();

    // ---- B fragments: K = hf*8..+7 then 16+hf*8..+7 (two contig chunks) ----
    v16bf bfr[4];
#pragma unroll
    for (int fn = 0; fn < 4; ++fn) {
      int cn = wn * 64 + fn * 16 + lr;
      v8bf blo = *(const v8bf*)&Bs[cn][hf * 8];
      v8bf bhi = *(const v8bf*)&Bs[cn][16 + hf * 8];
      bfr[fn] = __builtin_shufflevector(blo, bhi,
          0, 1, 2, 3, 4, 5, 6, 7, 8, 9, 10, 11, 12, 13, 14, 15);
    }
    // ---- A fragments: K = hf*8..+7 then 16+hf*8..+7 (two contig chunks) ----
#pragma unroll
    for (int fm = 0; fm < 2; ++fm) {
      int rm = wm * 32 + fm * 16 + lr;
      v8bf alo = *(const v8bf*)&As[rm][hf * 8];
      v8bf ahi = *(const v8bf*)&As[rm][16 + hf * 8];
      v16bf af = __builtin_shufflevector(alo, ahi,
          0, 1, 2, 3, 4, 5, 6, 7, 8, 9, 10, 11, 12, 13, 14, 15);
#pragma unroll
      for (int fn = 0; fn < 4; ++fn) {
        acc[fm][fn] = __builtin_amdgcn_wmma_f32_16x16x32_bf16(
            false, af, false, bfr[fn], (short)0, acc[fm][fn], false, false);
      }
    }
    __syncthreads();
  }

  // ---- Epilogue: C/D layout = VGPR r holds (M = r + hf*8, N = lr) ----
#pragma unroll
  for (int fm = 0; fm < 2; ++fm) {
#pragma unroll
    for (int fn = 0; fn < 4; ++fn) {
#pragma unroll
      for (int r = 0; r < 8; ++r) {
        int gm = m0 + wm * 32 + fm * 16 + hf * 8 + r;
        int gn = n0 + wn * 64 + fn * 16 + lr;
        if (gm < M && gn < Nd) {
          float y = acc[fm][fn][r];
          if (act == ACT_DIST) {
            y = rowadd[gm] + bias[gn] - 2.0f * y;
          } else {
            if (bias) y += bias[gn];
            if (bng)  y = y * (bng[gn] * BN_RSQ) + bnb[gn];
            if (act == ACT_LRELU)     y = (y >= 0.0f) ? y : alpha * y;
            else if (act == ACT_TANH) y = tanhf(y);
          }
          Y[(size_t)gm * ldy + gn] = y;
        }
      }
    }
  }
}

// ---------------------------------------------------------------------------
// Helper kernels
// ---------------------------------------------------------------------------

__global__ void style_in_kernel(const float* __restrict__ x,
                                const float* __restrict__ z,
                                float* __restrict__ S, long rows, int CX, int CZ)
{
  const int C = CX + CZ;
  long t = (long)blockIdx.x * blockDim.x + threadIdx.x;
  if (t >= rows * C) return;
  int c = (int)(t % C);
  long r = t / C;
  S[t] = (c < CX) ? x[r * CX + c] : z[r * CZ + (c - CX)];
}

__global__ void sqnorm_kernel(const float* __restrict__ X, float* __restrict__ sq,
                              long rows, int C)
{
  long r = (long)blockIdx.x * blockDim.x + threadIdx.x;
  if (r >= rows) return;
  float s = 0.0f;
  for (int c = 0; c < C; ++c) { float v = X[r * C + c]; s += v * v; }
  sq[r] = s;
}

__global__ void knn_topk_kernel(const float* __restrict__ dist,
                                int* __restrict__ idx, int N)
{
  int n = blockIdx.x * blockDim.x + threadIdx.x;
  if (n >= N) return;
  float bd[KNN]; int bi[KNN];
#pragma unroll
  for (int i = 0; i < KNN; ++i) { bd[i] = 3.4e38f; bi[i] = 0; }
  const float* row = dist + (size_t)n * N;
  for (int m = 0; m < N; ++m) {
    if (m == n) continue;           // argsort[:,1:] skips self
    float d = row[m];
    if (d < bd[KNN - 1]) {
      int j = KNN - 1;
      while (j > 0 && bd[j - 1] > d) { bd[j] = bd[j - 1]; bi[j] = bi[j - 1]; --j; }
      bd[j] = d; bi[j] = m;
    }
  }
  for (int i = 0; i < KNN; ++i) idx[(size_t)n * KNN + i] = bi[i];
}

// e[(n*KNN+k)][c] : c<C -> central ; c>=C -> neigh - central
__global__ void edge_build_kernel(const float* __restrict__ Xb,
                                  const int* __restrict__ idxb,
                                  float* __restrict__ e, int N, int C)
{
  const int C2 = 2 * C;
  long total = (long)N * KNN * C2;
  long t = (long)blockIdx.x * blockDim.x + threadIdx.x;
  if (t >= total) return;
  int c = (int)(t % C2);
  long row = t / C2;
  int k = (int)(row % KNN);
  long n = row / KNN;
  int cc = (c < C) ? c : c - C;
  float central = Xb[n * C + cc];
  float v = central;
  if (c >= C) {
    int j = idxb[n * KNN + k];
    v = Xb[(long)j * C + cc] - central;
  }
  e[row * C2 + c] = v;
}

// softmax over k of W[(n*KNN+k)][c]
__global__ void softmax_k_kernel(float* __restrict__ W, int N, int F)
{
  long total = (long)N * F;
  long t = (long)blockIdx.x * blockDim.x + threadIdx.x;
  if (t >= total) return;
  int c = (int)(t % F);
  long n = t / F;
  float* base = W + n * (long)KNN * F + c;
  float mx = -3.4e38f;
#pragma unroll
  for (int k = 0; k < KNN; ++k) mx = fmaxf(mx, base[(long)k * F]);
  float e[KNN]; float s = 0.0f;
#pragma unroll
  for (int k = 0; k < KNN; ++k) { e[k] = __expf(base[(long)k * F] - mx); s += e[k]; }
  float inv = 1.0f / s;
#pragma unroll
  for (int k = 0; k < KNN; ++k) base[(long)k * F] = e[k] * inv;
}

__global__ void elemmul_kernel(float* __restrict__ a, const float* __restrict__ b,
                               long total)
{
  long t = (long)blockIdx.x * blockDim.x + threadIdx.x;
  if (t < total) a[t] *= b[t];
}

// ow[o][c][k] -> owt[o][k*F + c]  (makes the bcnk,ock->bon einsum a GEMM)
__global__ void repack_ow_kernel(const float* __restrict__ ow,
                                 float* __restrict__ owt, int F)
{
  long total = (long)F * F * KNN;
  long t = (long)blockIdx.x * blockDim.x + threadIdx.x;
  if (t >= total) return;
  int k = (int)(t % KNN);
  long rem = t / KNN;
  int c = (int)(rem % F);
  long o = rem / F;
  owt[o * (long)KNN * F + (long)k * F + c] = ow[t];
}

// X = sty[:, :F] * X + sty[:, F:]
__global__ void adain_kernel(float* __restrict__ X, const float* __restrict__ sty,
                             long rows, int F)
{
  long total = rows * F;
  long t = (long)blockIdx.x * blockDim.x + threadIdx.x;
  if (t >= total) return;
  int c = (int)(t % F);
  long r = t / F;
  float g  = sty[r * (long)(2 * F) + c];
  float be = sty[r * (long)(2 * F) + F + c];
  X[t] = g * X[t] + be;
}

__global__ void gmax_kernel(const float* __restrict__ X3, float* __restrict__ fg,
                            int Bv, int N, int F)
{
  int t = blockIdx.x * blockDim.x + threadIdx.x;
  if (t >= Bv * F) return;
  int c = t % F, b = t / F;
  float m = -3.4e38f;
  for (int n = 0; n < N; ++n)
    m = fmaxf(m, X3[((long)b * N + n) * F + c]);
  fg[t] = m;
}

// feat[(b*N+n)][c] : c<F -> fg[b][c] ; else x3[(b*N+n)][c-F]
__global__ void feat_kernel(const float* __restrict__ fg, const float* __restrict__ x3,
                            float* __restrict__ feat, int Bv, int N, int F)
{
  long total = (long)Bv * N * 2 * F;
  long t = (long)blockIdx.x * blockDim.x + threadIdx.x;
  if (t >= total) return;
  int c = (int)(t % (2 * F));
  long r = t / (2 * F);
  int b = (int)(r / N);
  feat[t] = (c < F) ? fg[(long)b * F + c] : x3[r * F + (c - F)];
}

// t4[(b*N+n)][c] -> out[b][c][n]
__global__ void out_transpose_kernel(const float* __restrict__ t4,
                                     float* __restrict__ out, int Bv, int N)
{
  long total = (long)Bv * N * 3;
  long t = (long)blockIdx.x * blockDim.x + threadIdx.x;
  if (t >= total) return;
  int c = (int)(t % 3);
  long r = t / 3;
  int n = (int)(r % N);
  int b = (int)(r / N);
  out[((long)b * 3 + c) * N + n] = t4[t];
}

// ---------------------------------------------------------------------------
// Host side
// ---------------------------------------------------------------------------

static inline dim3 grid1(long total, int tb) {
  return dim3((unsigned)((total + tb - 1) / tb));
}

static void launch_gemm(hipStream_t st, const float* A, int lda,
                        const float* W, int ldw, float* Y, int ldy,
                        int M, int Kd, int Nd,
                        const float* bias, const float* bng, const float* bnb,
                        const float* rowadd, int act, float alpha)
{
  dim3 grid((Nd + TBN - 1) / TBN, (M + TBM - 1) / TBM);
  gemm_bf16_wmma<<<grid, 256, 0, st>>>(A, lda, W, ldw, Y, ldy, M, Kd, Nd,
                                       bias, bng, bnb, rowadd, act, alpha);
}

extern "C" void kernel_launch(void* const* d_in, const int* in_sizes, int n_in,
                              void* d_out, int out_size, void* d_ws, size_t ws_size,
                              hipStream_t stream)
{
  (void)in_sizes; (void)n_in; (void)out_size; (void)ws_size;
  const int  Bv = 4, N = 2048, NZ = 128;
  const long RN = (long)Bv * N;    // 8192 rows

  auto F32 = [&](int i) { return (const float*)d_in[i]; };

  // bump allocator over d_ws
  char* cur = (char*)d_ws;
  auto allocf = [&](size_t elems) -> float* {
    float* p = (float*)cur;
    cur += ((elems * sizeof(float) + 255) & ~(size_t)255);
    return p;
  };
  auto alloci = [&](size_t elems) -> int* {
    int* p = (int*)cur;
    cur += ((elems * sizeof(int) + 255) & ~(size_t)255);
    return p;
  };

  float* styin = allocf(RN * 131);
  float* s1    = allocf(RN * 1024);
  float* S     = allocf(RN * 1024);
  float* sq    = allocf(RN);
  float* dist  = allocf((size_t)N * N);          // reused per batch
  int*   idx   = alloci(RN * KNN);
  float* ebuf  = allocf((size_t)N * KNN * 1024); // max 2C = 1024 (e3)
  float* wb1   = allocf((size_t)N * KNN * 512);
  float* wb2   = allocf((size_t)N * KNN * 1024);
  float* hb    = allocf((size_t)N * KNN * 1024);
  float* owt   = allocf((size_t)1024 * 1024 * KNN);
  float* sty   = allocf(RN * 2048);
  float* x1b_  = allocf(RN * 128);
  float* x2b_  = allocf(RN * 512);
  float* x3b_  = allocf(RN * 1024);
  float* fg0   = allocf((size_t)Bv * 1024);
  float* fg1   = allocf((size_t)Bv * 1024);
  float* fg2   = allocf((size_t)Bv * 1024);
  float* feat  = allocf(RN * 2048);
  float* tb1   = allocf(RN * 512);
  float* tb2   = allocf(RN * 128);
  float* tb3   = allocf(RN * 32);
  float* tb4   = allocf(RN * 3);

  // ---- style MLP: s = lrelu(W2 lrelu(W1 [x;z] + b1) + b2) ----
  style_in_kernel<<<grid1(RN * 131, 256), 256, 0, stream>>>(
      F32(0), F32(1), styin, RN, 3, NZ);
  launch_gemm(stream, styin, 131, F32(2), 131, s1, 1024, (int)RN, 131, 1024,
              F32(3), nullptr, nullptr, nullptr, ACT_LRELU, 0.01f);
  launch_gemm(stream, s1, 1024, F32(4), 1024, S, 1024, (int)RN, 1024, 1024,
              F32(5), nullptr, nullptr, nullptr, ACT_LRELU, 0.01f);

  // ---- one edge block + adain ----
  auto edge_stage = [&](const float* Xin, int C, int Fout, int hdim,
                        int pbase, int abase, float* Xout)
  {
    const float* w1   = F32(pbase + 0);
    const float* w1b  = F32(pbase + 1);
    const float* bn1g = F32(pbase + 2);
    const float* bn1b = F32(pbase + 3);
    const float* w2   = F32(pbase + 4);
    const float* w2b  = F32(pbase + 5);
    const float* bn2g = F32(pbase + 6);
    const float* bn2b = F32(pbase + 7);
    const float* x1w  = F32(pbase + 8);
    const float* x1bb = F32(pbase + 9);
    const float* bnxg = F32(pbase + 10);
    const float* bnxb = F32(pbase + 11);
    const float* ow   = F32(pbase + 12);
    const float* ob   = F32(pbase + 13);

    sqnorm_kernel<<<grid1(RN, 256), 256, 0, stream>>>(Xin, sq, RN, C);
    repack_ow_kernel<<<grid1((long)Fout * Fout * KNN, 256), 256, 0, stream>>>(
        ow, owt, Fout);

    const long MK = (long)N * KNN;
    for (int b = 0; b < Bv; ++b) {
      const float* Xb = Xin + (long)b * N * C;
      // pairwise sq distances: Gram via WMMA + fused dist epilogue
      launch_gemm(stream, Xb, C, Xb, C, dist, N, N, C, N,
                  sq + (long)b * N, nullptr, nullptr, sq + (long)b * N,
                  ACT_DIST, 0.0f);
      knn_topk_kernel<<<grid1(N, 128), 128, 0, stream>>>(
          dist, idx + (long)b * N * KNN, N);
      edge_build_kernel<<<grid1(MK * 2 * C, 256), 256, 0, stream>>>(
          Xb, idx + (long)b * N * KNN, ebuf, N, C);
      // w-branch: two convs on e[:, C:]
      launch_gemm(stream, ebuf + C, 2 * C, w1, C, wb1, hdim,
                  (int)MK, C, hdim, w1b, bn1g, bn1b, nullptr, ACT_LRELU, 0.01f);
      launch_gemm(stream, wb1, hdim, w2, hdim, wb2, Fout,
                  (int)MK, hdim, Fout, w2b, bn2g, bn2b, nullptr, ACT_LRELU, 0.01f);
      softmax_k_kernel<<<grid1((long)N * Fout, 256), 256, 0, stream>>>(
          wb2, N, Fout);
      // h-branch conv on full e
      launch_gemm(stream, ebuf, 2 * C, x1w, 2 * C, hb, Fout,
                  (int)MK, 2 * C, Fout, x1bb, bnxg, bnxb, nullptr, ACT_LRELU, 0.01f);
      elemmul_kernel<<<grid1(MK * Fout, 256), 256, 0, stream>>>(
          hb, wb2, MK * Fout);
      // einsum bcnk,ock->bon as GEMM over flattened (k,c); + lrelu(0.2)
      launch_gemm(stream, hb, KNN * Fout, owt, KNN * Fout,
                  Xout + (long)b * N * Fout, Fout,
                  N, KNN * Fout, Fout, ob, nullptr, nullptr, nullptr,
                  ACT_LRELU, 0.2f);
    }
    // adain: sty = S * sw^T + sb ; X = g*X + be
    launch_gemm(stream, S, 1024, F32(abase + 0), 1024, sty, 2 * Fout,
                (int)RN, 1024, 2 * Fout, F32(abase + 1),
                nullptr, nullptr, nullptr, ACT_NONE, 0.0f);
    adain_kernel<<<grid1(RN * Fout, 256), 256, 0, stream>>>(Xout, sty, RN, Fout);
  };

  edge_stage(F32(0), 3,   128, 64,  6,  20, x1b_);   // e1 + a1
  edge_stage(x1b_,  128,  512, 256, 22, 36, x2b_);   // e2 + a2
  edge_stage(x2b_,  512, 1024, 512, 38, 52, x3b_);   // e3 + a3

  // ---- global features: max over n, two FCs with BN ----
  gmax_kernel<<<grid1((long)Bv * 1024, 256), 256, 0, stream>>>(
      x3b_, fg0, Bv, N, 1024);
  launch_gemm(stream, fg0, 1024, F32(54), 1024, fg1, 1024, Bv, 1024, 1024,
              F32(55), F32(56), F32(57), nullptr, ACT_LRELU, 0.01f);
  launch_gemm(stream, fg1, 1024, F32(58), 1024, fg2, 1024, Bv, 1024, 1024,
              F32(59), F32(60), F32(61), nullptr, ACT_LRELU, 0.01f);

  // ---- tail convs ----
  feat_kernel<<<grid1(RN * 2048, 256), 256, 0, stream>>>(
      fg2, x3b_, feat, Bv, N, 1024);
  launch_gemm(stream, feat, 2048, F32(62), 2048, tb1, 512, (int)RN, 2048, 512,
              F32(63), nullptr, nullptr, nullptr, ACT_LRELU, 0.01f);
  launch_gemm(stream, tb1, 512, F32(64), 512, tb2, 128, (int)RN, 512, 128,
              F32(65), nullptr, nullptr, nullptr, ACT_LRELU, 0.01f);
  launch_gemm(stream, tb2, 128, F32(66), 128, tb3, 32, (int)RN, 128, 32,
              F32(67), nullptr, nullptr, nullptr, ACT_LRELU, 0.01f);
  launch_gemm(stream, tb3, 32, F32(68), 32, tb4, 3, (int)RN, 32, 3,
              F32(69), nullptr, nullptr, nullptr, ACT_TANH, 0.0f);

  out_transpose_kernel<<<grid1(RN * 3, 256), 256, 0, stream>>>(
      tb4, (float*)d_out, Bv, N);
}
